// LSTMModel_26594437497058
// MI455X (gfx1250) — matrix operand order repaired and checked by reading, not compiled, not run on hardware
//
#include <hip/hip_runtime.h>
#include <hip/hip_bf16.h>
#include <math.h>

typedef __bf16 bf16;
typedef __attribute__((ext_vector_type(16))) __bf16 v16bf;
typedef __attribute__((ext_vector_type(8)))  __bf16 v8bf;
typedef __attribute__((ext_vector_type(8)))  float  v8f;

#define B_SZ 256
#define T_SZ 512
#define D_IN 32
#define H_SZ 256

#define NGROUP 8     // batch groups of 32 rows
#define WPG    8     // workgroups per group (gate-column split)
#define MWG    32    // batch rows per workgroup
#define NLOC   128   // gate columns per workgroup (32 hidden units x 4 gates)
#define KC0    9     // K chunks layer0: [x(32) | hA(256)] = 288
#define KC1    16    // K chunks layer1: [hA(256) | hB(256)] = 512
#define K0     288
#define K1     512
#define NTHREADS 256

#define W0_ELEMS (KC0*8*512)    // 36864 bf16
#define W1_ELEMS (KC1*8*512)    // 65536 bf16
#define A0_ELEMS (MWG*K0)       // 9216 bf16
#define A1_ELEMS (MWG*K1)       // 16384 bf16
#define GATES_ELEMS (MWG*NLOC)  // 4096 f32
#define SMEM_BYTES (((size_t)W0_ELEMS + W1_ELEMS + A0_ELEMS + A1_ELEMS)*2 + ((size_t)GATES_ELEMS + 2*NLOC)*4)

static __device__ __forceinline__ unsigned pack2bf(float a, float b) {
  unsigned short ua = __builtin_bit_cast(unsigned short, (__bf16)a);
  unsigned short ub = __builtin_bit_cast(unsigned short, (__bf16)b);
  return (unsigned)ua | ((unsigned)ub << 16);
}

static __device__ __forceinline__ float sigmoidf_(float x) {
  return 1.0f / (1.0f + __expf(-x));
}

// A-fragment (16x32 bf16, M x K) per ISA 7.12.2: lanes 0-15 hold K {kb..kb+7, kb+16..kb+23}
// with kb=kc*32; lanes 16-31 same with kb+8.
static __device__ __forceinline__ v16bf load_a_frag(const bf16* base, int rowstride,
                                                    int m0, int kc, int lane) {
  int m  = m0 + (lane & 15);
  int kb = kc * 32 + ((lane & 16) ? 8 : 0);
  const bf16* p = base + m * rowstride + kb;
  v8bf lo = *(const v8bf*)p;          // K = kb .. kb+7
  v8bf hi = *(const v8bf*)(p + 16);   // K = kb+16 .. kb+23
  v16bf a;
#pragma unroll
  for (int i = 0; i < 8; ++i) { a[i] = lo[i]; a[i + 8] = hi[i]; }
  return a;
}

// Monotonic-counter spin barrier across the 8 WGs of one batch group.
static __device__ __forceinline__ void group_spin_barrier(unsigned* ctr, unsigned target) {
  __hip_atomic_fetch_add(ctr, 1u, __ATOMIC_RELEASE, __HIP_MEMORY_SCOPE_AGENT);
  while (__hip_atomic_load(ctr, __ATOMIC_ACQUIRE, __HIP_MEMORY_SCOPE_AGENT) < target)
    __builtin_amdgcn_s_sleep(2);
}

__global__ __launch_bounds__(NTHREADS, 1)
void lstm2_fused(const float* __restrict__ x,
                 const float* __restrict__ W_ih0, const float* __restrict__ W_hh0,
                 const float* __restrict__ b0,
                 const float* __restrict__ W_ih1, const float* __restrict__ W_hh1,
                 const float* __restrict__ b1,
                 unsigned short* __restrict__ hAbuf,   // [256][256] bf16 bits
                 unsigned short* __restrict__ hBbuf,   // [256][256] bf16 bits
                 unsigned* __restrict__ ctrs)
{
  extern __shared__ char smem[];
  bf16*  W0f    = (bf16*)smem;                 // B-fragments, layer0 (weight-stationary)
  bf16*  W1f    = W0f + W0_ELEMS;              // B-fragments, layer1
  bf16*  A0     = W1f + W1_ELEMS;              // [32][288] = [x_t | hA_{t-1}]
  bf16*  A1     = A0 + A0_ELEMS;               // [32][512] = [hA_t | hB_{t-1}]
  float* gatesL = (float*)(A1 + A1_ELEMS);     // [32][128] gate exchange
  float* bias0  = gatesL + GATES_ELEMS;        // [128]
  float* bias1  = bias0 + NLOC;                // [128]

  const int tid   = threadIdx.x;
  const int lane  = tid & 31;
  const int w     = tid >> 5;          // wave id = local N-tile
  const int group = blockIdx.x >> 3;   // batch group
  const int q     = blockIdx.x & 7;    // hidden-unit slice within group
  const int mbase = group * MWG;
  unsigned* ctr   = ctrs + group * 16; // 64B-spaced counters

  // ---- one-time: weights -> bf16 B-fragment layout in LDS ----
  for (int e = tid; e < W0_ELEMS; e += NTHREADS) {
    int f = e >> 9, r = e & 511;
    int L = r >> 4, i = r & 15;
    int kc = f >> 3, nt = f & 7;
    int K  = kc*32 + ((L & 16) ? 8 : 0) + (i & 7) + ((i & 8) ? 16 : 0);  // B: lanes<16 K 0-7,16-23; lanes>=16 K 8-15,24-31
    int cl = nt*16 + (L & 15);                     // local gate column
    int g  = (cl >> 5)*H_SZ + q*32 + (cl & 31);    // global gate row of W
    float wv = (K < D_IN) ? W_ih0[g*D_IN + K] : W_hh0[g*H_SZ + (K - D_IN)];
    W0f[e] = (bf16)wv;
  }
  for (int e = tid; e < W1_ELEMS; e += NTHREADS) {
    int f = e >> 9, r = e & 511;
    int L = r >> 4, i = r & 15;
    int kc = f >> 3, nt = f & 7;
    int K  = kc*32 + ((L & 16) ? 8 : 0) + (i & 7) + ((i & 8) ? 16 : 0);
    int cl = nt*16 + (L & 15);
    int g  = (cl >> 5)*H_SZ + q*32 + (cl & 31);
    float wv = (K < H_SZ) ? W_ih1[g*H_SZ + K] : W_hh1[g*H_SZ + (K - H_SZ)];
    W1f[e] = (bf16)wv;
  }
  for (int c = tid; c < NLOC; c += NTHREADS) {
    int g = (c >> 5)*H_SZ + q*32 + (c & 31);
    bias0[c] = b0[g];
    bias1[c] = b1[g];
  }
  for (int e = tid; e < A0_ELEMS; e += NTHREADS) A0[e] = (bf16)0.0f;  // hA_{-1} = 0
  for (int e = tid; e < A1_ELEMS; e += NTHREADS) A1[e] = (bf16)0.0f;  // hB_{-1} = 0
  __syncthreads();

  float cA[4] = {0.f, 0.f, 0.f, 0.f};
  float cB[4] = {0.f, 0.f, 0.f, 0.f};
  const int em  = tid >> 3;          // batch row for elementwise
  const int ej0 = (tid & 7) << 2;    // first of 4 hidden units
  unsigned barnum = 0;

  for (int t = 0; t < T_SZ; ++t) {
    // ---- stage 1: x_t -> A0[:, 0:32] (f32 -> bf16) ----
    {
      int r = tid >> 3, cc = (tid & 7) << 2;
      const float* xp = x + ((size_t)(mbase + r) * T_SZ + t) * D_IN + cc;
      float4 xv = *(const float4*)xp;
      uint2 pk; pk.x = pack2bf(xv.x, xv.y); pk.y = pack2bf(xv.z, xv.w);
      *(uint2*)(A0 + r*K0 + cc) = pk;
      if (t + 1 < T_SZ) __builtin_prefetch(xp + D_IN, 0, 1);  // global_prefetch_b8
    }
    __syncthreads();

    // ---- layer 0 GEMM: gates0 = [x_t | hA] x W0slice ----
    {
      v8f acc0 = {0,0,0,0,0,0,0,0};
      v8f acc1 = {0,0,0,0,0,0,0,0};
#pragma unroll
      for (int kc = 0; kc < KC0; ++kc) {
        v16bf a0 = load_a_frag(A0, K0, 0,  kc, lane);
        v16bf a1 = load_a_frag(A0, K0, 16, kc, lane);
        v16bf bb = *(const v16bf*)(W0f + (kc*8 + w)*512 + lane*16);
        acc0 = __builtin_amdgcn_wmma_f32_16x16x32_bf16(false, a0, false, bb, (short)0, acc0, false, false);
        acc1 = __builtin_amdgcn_wmma_f32_16x16x32_bf16(false, a1, false, bb, (short)0, acc1, false, false);
      }
      int n = lane & 15, mh = (lane & 16) ? 8 : 0;
#pragma unroll
      for (int r = 0; r < 8; ++r) {
        gatesL[(mh + r)*NLOC      + w*16 + n] = acc0[r];
        gatesL[(mh + r + 16)*NLOC + w*16 + n] = acc1[r];
      }
    }
    __syncthreads();

    // ---- layer 0 elementwise: c/h update, hA slice -> global ----
    {
      float hv[4];
#pragma unroll
      for (int u = 0; u < 4; ++u) {
        int j = ej0 + u;
        float iv = sigmoidf_(gatesL[em*NLOC +       j] + bias0[j]);
        float fv = sigmoidf_(gatesL[em*NLOC + 32 +  j] + bias0[32 + j]);
        float gv = tanhf   (gatesL[em*NLOC + 64 +  j] + bias0[64 + j]);
        float ov = sigmoidf_(gatesL[em*NLOC + 96 +  j] + bias0[96 + j]);
        cA[u] = fv * cA[u] + iv * gv;
        hv[u] = ov * tanhf(cA[u]);
      }
      uint2 pk; pk.x = pack2bf(hv[0], hv[1]); pk.y = pack2bf(hv[2], hv[3]);
      *(uint2*)(hAbuf + (size_t)(mbase + em)*H_SZ + q*32 + ej0) = pk;
    }
    __threadfence();
    __syncthreads();
    barnum++;
    if (tid == 0) group_spin_barrier(ctr, WPG * barnum);
    __syncthreads();

    // ---- stage 3: gather full hA_t -> A1[:,0:256] & A0[:,32:288]; hB_{t-1} -> A1[:,256:512]
    for (int e = tid * 8; e < MWG * H_SZ; e += NTHREADS * 8) {
      uint4 v = *(const uint4*)(hAbuf + (size_t)mbase*H_SZ + e);
      int row = e >> 8, col = e & 255;
      *(uint4*)(A1 + row*K1 + col)      = v;
      *(uint4*)(A0 + row*K0 + 32 + col) = v;
    }
    if (t > 0) {
      for (int e = tid * 8; e < MWG * H_SZ; e += NTHREADS * 8) {
        uint4 v = *(const uint4*)(hBbuf + (size_t)mbase*H_SZ + e);
        int row = e >> 8, col = e & 255;
        *(uint4*)(A1 + row*K1 + 256 + col) = v;
      }
    }
    __syncthreads();

    // ---- layer 1 GEMM: gates1 = [hA_t | hB_{t-1}] x W1slice ----
    {
      v8f acc0 = {0,0,0,0,0,0,0,0};
      v8f acc1 = {0,0,0,0,0,0,0,0};
#pragma unroll
      for (int kc = 0; kc < KC1; ++kc) {
        v16bf a0 = load_a_frag(A1, K1, 0,  kc, lane);
        v16bf a1 = load_a_frag(A1, K1, 16, kc, lane);
        v16bf bb = *(const v16bf*)(W1f + (kc*8 + w)*512 + lane*16);
        acc0 = __builtin_amdgcn_wmma_f32_16x16x32_bf16(false, a0, false, bb, (short)0, acc0, false, false);
        acc1 = __builtin_amdgcn_wmma_f32_16x16x32_bf16(false, a1, false, bb, (short)0, acc1, false, false);
      }
      int n = lane & 15, mh = (lane & 16) ? 8 : 0;
#pragma unroll
      for (int r = 0; r < 8; ++r) {
        gatesL[(mh + r)*NLOC      + w*16 + n] = acc0[r];
        gatesL[(mh + r + 16)*NLOC + w*16 + n] = acc1[r];
      }
    }
    __syncthreads();

    // ---- layer 1 elementwise -> hB slice -> global ----
    {
      float hv[4];
#pragma unroll
      for (int u = 0; u < 4; ++u) {
        int j = ej0 + u;
        float iv = sigmoidf_(gatesL[em*NLOC +       j] + bias1[j]);
        float fv = sigmoidf_(gatesL[em*NLOC + 32 +  j] + bias1[32 + j]);
        float gv = tanhf   (gatesL[em*NLOC + 64 +  j] + bias1[64 + j]);
        float ov = sigmoidf_(gatesL[em*NLOC + 96 +  j] + bias1[96 + j]);
        cB[u] = fv * cB[u] + iv * gv;
        hv[u] = ov * tanhf(cB[u]);
      }
      uint2 pk; pk.x = pack2bf(hv[0], hv[1]); pk.y = pack2bf(hv[2], hv[3]);
      *(uint2*)(hBbuf + (size_t)(mbase + em)*H_SZ + q*32 + ej0) = pk;
    }
    __threadfence();
    __syncthreads();
    barnum++;
    if (tid == 0) group_spin_barrier(ctr, WPG * barnum);
    __syncthreads();
  }
}

__global__ void ws_init(unsigned* __restrict__ ctrs) {
  int i = threadIdx.x;
  if (i < NGROUP * 16) ctrs[i] = 0u;
}

__global__ void fc_last(const unsigned short* __restrict__ hBbuf,
                        const float* __restrict__ W_fc, const float* __restrict__ b_fc,
                        float* __restrict__ out) {
  int b = threadIdx.x;  // 256 threads
  float s = b_fc[0];
#pragma unroll 8
  for (int j = 0; j < H_SZ; ++j)
    s += (float)__builtin_bit_cast(__bf16, hBbuf[(size_t)b * H_SZ + j]) * W_fc[j];
  out[b] = s;
}

extern "C" void kernel_launch(void* const* d_in, const int* in_sizes, int n_in,
                              void* d_out, int out_size, void* d_ws, size_t ws_size,
                              hipStream_t stream) {
  (void)in_sizes; (void)n_in; (void)out_size; (void)ws_size;
  const float* x     = (const float*)d_in[0];
  const float* W_ih0 = (const float*)d_in[1];
  const float* W_hh0 = (const float*)d_in[2];
  const float* b0    = (const float*)d_in[3];
  const float* W_ih1 = (const float*)d_in[4];
  const float* W_hh1 = (const float*)d_in[5];
  const float* b1    = (const float*)d_in[6];
  const float* W_fc  = (const float*)d_in[7];
  const float* b_fc  = (const float*)d_in[8];

  unsigned short* hA = (unsigned short*)d_ws;                 // 128KB
  unsigned short* hB = hA + (size_t)B_SZ * H_SZ;              // 128KB
  unsigned* ctrs     = (unsigned*)(hB + (size_t)B_SZ * H_SZ); // 512B

  hipLaunchKernelGGL(ws_init, dim3(1), dim3(128), 0, stream, ctrs);
  hipLaunchKernelGGL(lstm2_fused, dim3(NGROUP * WPG), dim3(NTHREADS), SMEM_BYTES, stream,
                     x, W_ih0, W_hh0, b0, W_ih1, W_hh1, b1, hA, hB, ctrs);
  hipLaunchKernelGGL(fc_last, dim3(1), dim3(256), 0, stream, hB, W_fc, b_fc, (float*)d_out);
}
